// InformerModel_15942918603182
// MI455X (gfx1250) — matrix operand order repaired and checked by reading, not compiled, not run on hardware
//
#include <hip/hip_runtime.h>
#include <hip/hip_bf16.h>
#include <math.h>

// ---- model dims ----
#define B_     2
#define L_     2048
#define INDIM  32
#define DM     256
#define NH     8
#define DFF    1024
#define NL     2
#define NC     8
#define DK     32
#define U_     38
#define LN_EPS 1e-5f

typedef __attribute__((ext_vector_type(16))) __bf16 v16bf;
typedef __attribute__((ext_vector_type(8)))  __bf16 v8bf;
typedef __attribute__((ext_vector_type(8)))  float   v8f;

union ABf { v16bf v; v8bf h[2]; };

__device__ inline v8f v8f_zero() {
  v8f z = {0.f, 0.f, 0.f, 0.f, 0.f, 0.f, 0.f, 0.f};
  return z;
}

__device__ inline v8f wmma_bf16(v16bf a, v16bf b, v8f c) {
  // D = A(16x32 bf16) x B(32x16 bf16) + C(16x16 f32)
  return __builtin_amdgcn_wmma_f32_16x16x32_bf16(false, a, false, b, (short)0, c,
                                                 false, false);
}

// monotone float -> uint key (order-preserving)
__device__ inline unsigned fkey(float f) {
  unsigned u = __float_as_uint(f);
  return (u & 0x80000000u) ? ~u : (u | 0x80000000u);
}

// wave-local LDS ordering: drain DS counter (CDNA5 split wait)
__device__ inline void lds_fence_wave() {
  asm volatile("s_wait_dscnt 0x0" ::: "memory");
}

// ---------------- weight transpose + cast: W[K][N] f32 -> Wt[N][K] bf16 ----------------
__global__ void wt_kernel(const float* __restrict__ W, __bf16* __restrict__ Wt,
                          int K, int N) {
  int idx = blockIdx.x * blockDim.x + threadIdx.x;
  if (idx >= K * N) return;
  int n = idx / K, k = idx % K;
  Wt[idx] = (__bf16)W[k * N + n];
}

// ---------------- embedding + positional encoding ----------------
__global__ void embed_kernel(const float* __restrict__ x, const float* __restrict__ ew,
                             const float* __restrict__ eb, float* __restrict__ hf,
                             __bf16* __restrict__ hb) {
  int idx = blockIdx.x * blockDim.x + threadIdx.x;
  if (idx >= B_ * L_ * DM) return;
  int d = idx % DM;
  int l = (idx / DM) % L_;
  int b = idx / (DM * L_);
  const float* xr = x + (size_t)(b * L_ + l) * INDIM;
  float acc = eb[d];
#pragma unroll
  for (int i = 0; i < INDIM; ++i) acc += xr[i] * ew[i * DM + d];
  acc *= 16.0f;  // sqrt(256)
  int j = d >> 1;
  float div = __expf((float)(2 * j) * (-9.210340371976184f / (float)DM));
  float ang = (float)l * div;
  acc += (d & 1) ? __cosf(ang) : __sinf(ang);
  hf[idx] = acc;
  hb[idx] = (__bf16)acc;
}

// ---------------- generic bf16 WMMA GEMM: C[M,N] = A[M,K] * Wt[N,K]^T + bias ----------------
// one wave computes a 16(M) x 64(N) tile; blockDim=256 (8 waves).
// K unrolled by 2 with ping-pong fragment buffers (K % 64 == 0): no register rotation,
// loads for one buffer overlap WMMAs consuming the other.
__global__ void __launch_bounds__(256)
gemm_bf16_kernel(const __bf16* __restrict__ A, const __bf16* __restrict__ Wt,
                 const float* __restrict__ bias, float* __restrict__ Cf,
                 __bf16* __restrict__ Cb, int M, int N, int K, int relu) {
  int wave = blockIdx.x * (blockDim.x >> 5) + (threadIdx.x >> 5);
  int lane = threadIdx.x & 31;
  int nt64 = N >> 6;
  int mt = wave / nt64;
  int nt = wave % nt64;
  int mbase = mt << 4;
  int nbase = nt << 6;
  if (mbase >= M) return;
  int half = lane >> 4;
  int l15 = lane & 15;

  v8f acc[4];
#pragma unroll
  for (int s = 0; s < 4; ++s) acc[s] = v8f_zero();

  const __bf16* parow = A + (size_t)(mbase + l15) * K + half * 16;
  const __bf16* pbrow[4];
#pragma unroll
  for (int s = 0; s < 4; ++s)
    pbrow[s] = Wt + (size_t)(nbase + s * 16 + l15) * K + half * 16;

  ABf a0, b0[4], a1, b1[4];
  // prologue: fill buffer 0 (kk = 0)
  a0.h[0] = *(const v8bf*)(parow);
  a0.h[1] = *(const v8bf*)(parow + 8);
#pragma unroll
  for (int s = 0; s < 4; ++s) {
    b0[s].h[0] = *(const v8bf*)(pbrow[s]);
    b0[s].h[1] = *(const v8bf*)(pbrow[s] + 8);
  }

  int steps = K >> 6;  // 64-wide double-steps
  for (int s64 = 0; s64 < steps; ++s64) {
    int kk = s64 << 6;
    // fill buffer 1 (kk+32) while buffer 0 computes
    {
      const __bf16* pa = parow + kk + 32;
      a1.h[0] = *(const v8bf*)pa;
      a1.h[1] = *(const v8bf*)(pa + 8);
#pragma unroll
      for (int s = 0; s < 4; ++s) {
        const __bf16* pb = pbrow[s] + kk + 32;
        b1[s].h[0] = *(const v8bf*)pb;
        b1[s].h[1] = *(const v8bf*)(pb + 8);
      }
    }
#pragma unroll
    for (int s = 0; s < 4; ++s) acc[s] = wmma_bf16(a0.v, b0[s].v, acc[s]);
    // refill buffer 0 (kk+64) while buffer 1 computes
    if (s64 + 1 < steps) {
      const __bf16* pa = parow + kk + 64;
      a0.h[0] = *(const v8bf*)pa;
      a0.h[1] = *(const v8bf*)(pa + 8);
#pragma unroll
      for (int s = 0; s < 4; ++s) {
        const __bf16* pb = pbrow[s] + kk + 64;
        b0[s].h[0] = *(const v8bf*)pb;
        b0[s].h[1] = *(const v8bf*)(pb + 8);
      }
    }
#pragma unroll
    for (int s = 0; s < 4; ++s) acc[s] = wmma_bf16(a1.v, b1[s].v, acc[s]);
  }

  int mrow = mbase + half * 8;
#pragma unroll
  for (int s = 0; s < 4; ++s) {
    int col = nbase + s * 16 + l15;
    float bv = bias[col];
#pragma unroll
    for (int r = 0; r < 8; ++r) {
      float v = acc[s][r] + bv;
      if (relu) v = fmaxf(v, 0.f);
      size_t o = (size_t)(mrow + r) * N + col;
      if (Cf) Cf[o] = v;
      if (Cb) Cb[o] = (__bf16)v;
    }
  }
}

// ---------------- split heads: q,k -> [BH][L][32] bf16 ; v -> vt [BH][32][L] bf16 ----------------
__global__ void qkv_rearrange_kernel(const __bf16* __restrict__ qb,
                                     const __bf16* __restrict__ kb,
                                     const __bf16* __restrict__ vb,
                                     __bf16* __restrict__ qh, __bf16* __restrict__ kh,
                                     __bf16* __restrict__ vt) {
  int idx = blockIdx.x * blockDim.x + threadIdx.x;
  if (idx >= B_ * L_ * DM) return;
  int c = idx % DM;
  int l = (idx / DM) % L_;
  int b = idx / (DM * L_);
  int hh = c >> 5, d = c & 31;
  int bh = b * NH + hh;
  qh[((size_t)bh * L_ + l) * DK + d] = qb[idx];
  kh[((size_t)bh * L_ + l) * DK + d] = kb[idx];
  vt[((size_t)bh * DK + d) * L_ + l] = vb[idx];
}

// ---------------- fused ProbSparse attention: one block = 16 query rows of one (b,h) --------------
// LDS: scores strip 16x2048 f32 (128KB, never hits HBM) + 8x256 scratch + 16 row sums.
// Top-k threshold = exact 4-pass radix select over monotone keys (bins live in the
// 8x256 scratch region, per-wave; phase C later reuses it for PV partial reduction).
__global__ void __launch_bounds__(256)
attn_kernel(const __bf16* __restrict__ qh, const __bf16* __restrict__ kh,
            const __bf16* __restrict__ vt, __bf16* __restrict__ ctx) {
  extern __shared__ float smem[];
  float* S = smem;                  // 16 * 2048
  float* red = smem + 16 * L_;      // 8 * 256 (radix bins in phase B, PV partials in C)
  float* rsum = red + 8 * 256;      // 16

  int mblk = blockIdx.x & (L_ / 16 - 1);
  int bh = blockIdx.x >> 7;
  int b = bh >> 3;
  int hh = bh & 7;
  int mbase = mblk << 4;

  int w = threadIdx.x >> 5;
  int lane = threadIdx.x & 31;
  int half = lane >> 4, l15 = lane & 15;
  const float scale = 0.17677669529663687f;  // 1/sqrt(DK)

  // ---- Phase A: S = Q K^T / sqrt(dk); one WMMA per 16x16 tile (whole d_k in one K-step).
  // 16 tiles per wave, ping-pong K-fragment buffers (even trip count, no reg rotation).
  ABf aq;
  {
    const __bf16* pq = qh + ((size_t)bh * L_ + (mbase + l15)) * DK + half * 16;
    aq.h[0] = *(const v8bf*)pq;
    aq.h[1] = *(const v8bf*)(pq + 8);
  }
  const __bf16* krow = kh + ((size_t)bh * L_ + l15) * DK + half * 16;
  ABf bk0, bk1;
  {
    const __bf16* pk = krow + (size_t)(w << 4) * DK;
    bk0.h[0] = *(const v8bf*)pk;
    bk0.h[1] = *(const v8bf*)(pk + 8);
  }
  for (int tt = 0; tt < 16; tt += 2) {
    int t0 = w + tt * 8;
    {
      const __bf16* pk = krow + (size_t)((t0 + 8) << 4) * DK;
      bk1.h[0] = *(const v8bf*)pk;
      bk1.h[1] = *(const v8bf*)(pk + 8);
    }
    v8f sacc = v8f_zero();
    sacc = wmma_bf16(aq.v, bk0.v, sacc);
    {
      int nb = t0 << 4;
      int row0 = half * 8;
#pragma unroll
      for (int r = 0; r < 8; ++r) S[(row0 + r) * L_ + nb + l15] = sacc[r] * scale;
    }
    if (tt + 2 < 16) {
      const __bf16* pk = krow + (size_t)((t0 + 16) << 4) * DK;
      bk0.h[0] = *(const v8bf*)pk;
      bk0.h[1] = *(const v8bf*)(pk + 8);
    }
    v8f sacc2 = v8f_zero();
    sacc2 = wmma_bf16(aq.v, bk1.v, sacc2);
    {
      int nb = (t0 + 8) << 4;
      int row0 = half * 8;
#pragma unroll
      for (int r = 0; r < 8; ++r) S[(row0 + r) * L_ + nb + l15] = sacc2[r] * scale;
    }
  }
  __syncthreads();

  // ---- Phase B: exact U-th-largest threshold via 256-bin radix select + masked softmax
  int* bins = (int*)(red + w * 256);  // per-wave 256 bins
  for (int rr = 0; rr < 2; ++rr) {
    int row = w * 2 + rr;
    float* Sr = S + row * L_;

    float mx = -3.4e38f;
    for (int j = lane; j < L_; j += 32) mx = fmaxf(mx, Sr[j]);
#pragma unroll
    for (int o = 16; o; o >>= 1) mx = fmaxf(mx, __shfl_xor(mx, o, 32));

    unsigned prefix = 0u;
    int need = U_;
#pragma unroll
    for (int pass = 0; pass < 4; ++pass) {
      int shift = 24 - pass * 8;
      // zero bins
      for (int i = lane; i < 256; i += 32) bins[i] = 0;
      lds_fence_wave();
      // histogram of candidates matching the resolved high-bit prefix
      unsigned pref_hi = (pass == 0) ? 0u : (prefix >> (shift + 8));
      for (int j = lane; j < L_; j += 32) {
        unsigned kk = fkey(Sr[j]);
        bool okx = (pass == 0) || ((kk >> (shift + 8)) == pref_hi);
        if (okx) atomicAdd(&bins[(kk >> shift) & 255], 1);
      }
      lds_fence_wave();
      // lane owns bytes [lane*8, lane*8+7]; suffix-sum across lanes (high bytes first)
      int s_lane = 0;
#pragma unroll
      for (int t = 0; t < 8; ++t) s_lane += bins[lane * 8 + t];
      int suf = s_lane;
#pragma unroll
      for (int o = 1; o < 32; o <<= 1) {
        int v = __shfl_down(suf, o, 32);
        if (lane + o < 32) suf += v;
      }
      int hi = suf - s_lane;  // count of bytes strictly above this lane's range
      unsigned long long bm = __ballot(hi < need && need <= suf);
      int lsel = __ffsll(bm) - 1;
      int hi_sel = __shfl(hi, lsel, 32);
      // refine within the selected lane's 8 bins (top byte downward)
      int base = lsel * 8;
      int cum = hi_sel, pick = base, nn = need;
#pragma unroll
      for (int t = 7; t >= 0; --t) {
        int c = bins[base + t];
        if (pick == base && cum < need && need <= cum + c) {
          pick = base + t;
          nn = need - cum;
        }
        cum += c;
      }
      need = nn;
      prefix |= ((unsigned)pick) << shift;
    }
    unsigned thr = prefix;  // exact U-th largest key

    float lsum = 0.f;
    for (int j = lane; j < L_; j += 32) {
      float s = Sr[j];
      float e = (fkey(s) >= thr) ? __expf(s - mx) : 0.f;
      Sr[j] = e;
      lsum += e;
    }
#pragma unroll
    for (int o = 16; o; o >>= 1) lsum += __shfl_xor(lsum, o, 32);
    if (lane == 0) rsum[row] = lsum;
  }
  __syncthreads();

  // ---- Phase C: ctx = P V ; 8 waves = 2 d-tiles x 4 K-chunks; ping-pong V fragments
  int dtile = w >> 2;  // 0..1
  int chunk = w & 3;   // 0..3
  v8f cacc = v8f_zero();
  const __bf16* vbase = vt + ((size_t)bh * DK + dtile * 16 + l15) * L_ + half * 16;
  int k0 = chunk * (L_ / 4);
  ABf bv0, bv1;
  {
    const __bf16* pv = vbase + k0;
    bv0.h[0] = *(const v8bf*)pv;
    bv0.h[1] = *(const v8bf*)(pv + 8);
  }
  const int csteps = (L_ / 4) >> 6;  // 8 double-steps
  for (int s64 = 0; s64 < csteps; ++s64) {
    int kk = k0 + (s64 << 6);
    {
      const __bf16* pv = vbase + kk + 32;
      bv1.h[0] = *(const v8bf*)pv;
      bv1.h[1] = *(const v8bf*)(pv + 8);
    }
    ABf ap;
#pragma unroll
    for (int e = 0; e < 8; ++e) {
      ap.v[e] = (__bf16)S[l15 * L_ + kk + half * 16 + e];
      ap.v[8 + e] = (__bf16)S[l15 * L_ + kk + 8 + half * 16 + e];
    }
    cacc = wmma_bf16(ap.v, bv0.v, cacc);
    if (s64 + 1 < csteps) {
      const __bf16* pv = vbase + kk + 64;
      bv0.h[0] = *(const v8bf*)pv;
      bv0.h[1] = *(const v8bf*)(pv + 8);
    }
    ABf ap2;
#pragma unroll
    for (int e = 0; e < 8; ++e) {
      ap2.v[e] = (__bf16)S[l15 * L_ + kk + 32 + half * 16 + e];
      ap2.v[8 + e] = (__bf16)S[l15 * L_ + kk + 40 + half * 16 + e];
    }
    cacc = wmma_bf16(ap2.v, bv1.v, cacc);
  }
  float* myred = red + w * 256;
#pragma unroll
  for (int r = 0; r < 8; ++r) myred[(half * 8 + r) * 16 + l15] = cacc[r];
  __syncthreads();

  for (int i = threadIdx.x; i < 512; i += 256) {
    int row = i >> 5, d = i & 31;
    int dt = d >> 4;
    float sum = 0.f;
#pragma unroll
    for (int c = 0; c < 4; ++c) sum += red[(dt * 4 + c) * 256 + row * 16 + (d & 15)];
    float val = sum / rsum[row];
    ctx[((size_t)(b * L_ + mbase + row)) * DM + hh * 32 + d] = (__bf16)val;
  }
}

// ---------------- fused residual add + LayerNorm (in-place on h) ----------------
__global__ void __launch_bounds__(DM)
ln_residual_kernel(float* __restrict__ hf, __bf16* __restrict__ hb,
                   const float* __restrict__ a, const float* __restrict__ g,
                   const float* __restrict__ bt) {
  __shared__ float red[DM];
  int row = blockIdx.x, tid = threadIdx.x;
  float s = hf[(size_t)row * DM + tid] + a[(size_t)row * DM + tid];
  red[tid] = s;
  __syncthreads();
  for (int o = DM / 2; o; o >>= 1) {
    if (tid < o) red[tid] += red[tid + o];
    __syncthreads();
  }
  float mean = red[0] / DM;
  __syncthreads();
  float d = s - mean;
  red[tid] = d * d;
  __syncthreads();
  for (int o = DM / 2; o; o >>= 1) {
    if (tid < o) red[tid] += red[tid + o];
    __syncthreads();
  }
  float var = red[0] / DM;
  float r = rsqrtf(var + LN_EPS);
  float o_ = d * r * g[tid] + bt[tid];
  hf[(size_t)row * DM + tid] = o_;
  hb[(size_t)row * DM + tid] = (__bf16)o_;
}

// ---------------- decoder: out[M,8] = h[M,256] @ dec_w + dec_b (N too small for WMMA) --------
__global__ void dec_kernel(const float* __restrict__ hf, const float* __restrict__ dw,
                           const float* __restrict__ db, float* __restrict__ out) {
  int idx = blockIdx.x * blockDim.x + threadIdx.x;
  if (idx >= B_ * L_ * NC) return;
  int c = idx & 7, row = idx >> 3;
  float acc = db[c];
  const float* hr = hf + (size_t)row * DM;
#pragma unroll 8
  for (int k = 0; k < DM; ++k) acc += hr[k] * dw[k * NC + c];
  out[idx] = acc;
}

// =====================================================================================
extern "C" void kernel_launch(void* const* d_in, const int* in_sizes, int n_in,
                              void* d_out, int out_size, void* d_ws, size_t ws_size,
                              hipStream_t stream) {
  const float* x     = (const float*)d_in[0];
  const float* emb_w = (const float*)d_in[1];
  const float* emb_b = (const float*)d_in[2];
  const float* wq    = (const float*)d_in[3];
  const float* bq    = (const float*)d_in[4];
  const float* wk    = (const float*)d_in[5];
  const float* bk    = (const float*)d_in[6];
  const float* wv    = (const float*)d_in[7];
  const float* bv    = (const float*)d_in[8];
  const float* wo    = (const float*)d_in[9];
  const float* bo    = (const float*)d_in[10];
  const float* ln1g  = (const float*)d_in[11];
  const float* ln1b  = (const float*)d_in[12];
  const float* ln2g  = (const float*)d_in[13];
  const float* ln2b  = (const float*)d_in[14];
  const float* ff1w  = (const float*)d_in[15];
  const float* ff1b  = (const float*)d_in[16];
  const float* ff2w  = (const float*)d_in[17];
  const float* ff2b  = (const float*)d_in[18];
  const float* decw  = (const float*)d_in[19];
  const float* decb  = (const float*)d_in[20];
  float* outp = (float*)d_out;

  const int M = B_ * L_;  // 4096

  // bump allocator over workspace
  char* ws = (char*)d_ws;
  size_t off = 0;
  auto alloc = [&](size_t bytes) -> void* {
    void* p = (void*)(ws + off);
    off += bytes;
    off = (off + 255) & ~(size_t)255;
    return p;
  };

  float*  h_f  = (float*)alloc((size_t)M * DM * 4);
  __bf16* h_b  = (__bf16*)alloc((size_t)M * DM * 2);
  __bf16* qb   = (__bf16*)alloc((size_t)M * DM * 2);
  __bf16* kb   = (__bf16*)alloc((size_t)M * DM * 2);
  __bf16* vb   = (__bf16*)alloc((size_t)M * DM * 2);
  __bf16* qhp  = (__bf16*)alloc((size_t)B_ * NH * L_ * DK * 2);
  __bf16* khp  = (__bf16*)alloc((size_t)B_ * NH * L_ * DK * 2);
  __bf16* vtp  = (__bf16*)alloc((size_t)B_ * NH * DK * L_ * 2);
  __bf16* ctxb = (__bf16*)alloc((size_t)M * DM * 2);
  float*  a_f  = (float*)alloc((size_t)M * DM * 4);
  __bf16* f1_b = (__bf16*)alloc((size_t)M * DFF * 2);
  __bf16* wqT[NL], *wkT[NL], *wvT[NL], *woT[NL], *f1T[NL], *f2T[NL];
  for (int l = 0; l < NL; ++l) {
    wqT[l] = (__bf16*)alloc((size_t)DM * DM * 2);
    wkT[l] = (__bf16*)alloc((size_t)DM * DM * 2);
    wvT[l] = (__bf16*)alloc((size_t)DM * DM * 2);
    woT[l] = (__bf16*)alloc((size_t)DM * DM * 2);
    f1T[l] = (__bf16*)alloc((size_t)DM * DFF * 2);
    f2T[l] = (__bf16*)alloc((size_t)DFF * DM * 2);
  }

  // weight prep (transpose + bf16 cast)
  for (int l = 0; l < NL; ++l) {
    int e1 = DM * DM, e2 = DM * DFF;
    wt_kernel<<<(e1 + 255) / 256, 256, 0, stream>>>(wq + (size_t)l * e1, wqT[l], DM, DM);
    wt_kernel<<<(e1 + 255) / 256, 256, 0, stream>>>(wk + (size_t)l * e1, wkT[l], DM, DM);
    wt_kernel<<<(e1 + 255) / 256, 256, 0, stream>>>(wv + (size_t)l * e1, wvT[l], DM, DM);
    wt_kernel<<<(e1 + 255) / 256, 256, 0, stream>>>(wo + (size_t)l * e1, woT[l], DM, DM);
    wt_kernel<<<(e2 + 255) / 256, 256, 0, stream>>>(ff1w + (size_t)l * e2, f1T[l], DM, DFF);
    wt_kernel<<<(e2 + 255) / 256, 256, 0, stream>>>(ff2w + (size_t)l * e2, f2T[l], DFF, DM);
  }

  // embedding + pos-encoding
  embed_kernel<<<(M * DM + 255) / 256, 256, 0, stream>>>(x, emb_w, emb_b, h_f, h_b);

  const size_t attn_shmem = (16 * L_ + 8 * 256 + 16) * sizeof(float);

  for (int l = 0; l < NL; ++l) {
    int blk_256 = (M / 16) * (DM / 64) / 8;    // 128
    int blk_1024 = (M / 16) * (DFF / 64) / 8;  // 512
    // QKV projections (bf16 out only)
    gemm_bf16_kernel<<<blk_256, 256, 0, stream>>>(h_b, wqT[l], bq + l * DM, nullptr, qb,
                                                  M, DM, DM, 0);
    gemm_bf16_kernel<<<blk_256, 256, 0, stream>>>(h_b, wkT[l], bk + l * DM, nullptr, kb,
                                                  M, DM, DM, 0);
    gemm_bf16_kernel<<<blk_256, 256, 0, stream>>>(h_b, wvT[l], bv + l * DM, nullptr, vb,
                                                  M, DM, DM, 0);
    // split heads, transpose V
    qkv_rearrange_kernel<<<(M * DM + 255) / 256, 256, 0, stream>>>(qb, kb, vb, qhp, khp,
                                                                   vtp);
    // fused ProbSparse attention
    attn_kernel<<<B_ * NH * (L_ / 16), 256, attn_shmem, stream>>>(qhp, khp, vtp, ctxb);
    // output projection (f32 out)
    gemm_bf16_kernel<<<blk_256, 256, 0, stream>>>(ctxb, woT[l], bo + l * DM, a_f, nullptr,
                                                  M, DM, DM, 0);
    // residual + LN1 (in-place h)
    ln_residual_kernel<<<M, DM, 0, stream>>>(h_f, h_b, a_f, ln1g + l * DM, ln1b + l * DM);
    // FF1 (+ReLU, bf16 out), FF2 (f32 out)
    gemm_bf16_kernel<<<blk_1024, 256, 0, stream>>>(h_b, f1T[l], ff1b + l * DFF, nullptr,
                                                   f1_b, M, DFF, DM, 1);
    gemm_bf16_kernel<<<blk_256, 256, 0, stream>>>(f1_b, f2T[l], ff2b + l * DM, a_f,
                                                  nullptr, M, DM, DFF, 0);
    // residual + LN2
    ln_residual_kernel<<<M, DM, 0, stream>>>(h_f, h_b, a_f, ln2g + l * DM, ln2b + l * DM);
  }

  // decoder head
  dec_kernel<<<(M * NC + 255) / 256, 256, 0, stream>>>(h_f, decw, decb, outp);
}
